// AlexCapsNet_Fully_MNIST_87488483820201
// MI455X (gfx1250) — matrix operand order repaired and checked by reading, not compiled, not run on hardware
//
#include <hip/hip_runtime.h>
#include <math.h>

// ---------------------------------------------------------------------------
// AlexCapsNet forward for MI455X (gfx1250), wave32 + WMMA bf16.
//
// Roofline: ~220 GFLOP total vs ~0.3 GB of traffic -> strongly compute bound,
// so all conv/FC layers are lowered to implicit-GEMM on
// V_WMMA_F32_16X16X32_BF16 (bf16 A/B, f32 accumulate). Weight tiles are staged
// to LDS with GLOBAL_LOAD_ASYNC_TO_LDS_B128 (ASYNCcnt) where shapes permit.
// All hot-loop indexing is 32-bit (every buffer < 2^31 elements).
// ---------------------------------------------------------------------------

#define BQ 512  // batch size from the reference

typedef __attribute__((ext_vector_type(16))) __bf16          v16bf;
typedef __attribute__((ext_vector_type(8)))  float           v8f;
typedef __attribute__((ext_vector_type(8)))  unsigned short  u16x8;
typedef __attribute__((ext_vector_type(16))) unsigned short  u16x16;

__device__ __forceinline__ unsigned short f2bf(float f) {
  unsigned u = __float_as_uint(f);
  u += 0x7FFFu + ((u >> 16) & 1u);           // round-to-nearest-even
  return (unsigned short)(u >> 16);
}
__device__ __forceinline__ float bf2f(unsigned short s) {
  return __uint_as_float(((unsigned)s) << 16);
}

// Build a 16-element bf16 fragment from two 16B-aligned LDS chunks without
// going through a union (keeps everything in VGPRs, no stack slot).
__device__ __forceinline__ v16bf ldsFrag(const unsigned short* p0,
                                         const unsigned short* p1) {
  const u16x8 lo = *(const u16x8*)p0;
  const u16x8 hi = *(const u16x8*)p1;
  const u16x16 w = __builtin_shufflevector(lo, hi, 0, 1, 2, 3, 4, 5, 6, 7,
                                           8, 9, 10, 11, 12, 13, 14, 15);
  return __builtin_bit_cast(v16bf, w);
}

// ---------------------------------------------------------------------------
// f32 -> bf16 bit conversion / zero fill
// ---------------------------------------------------------------------------
__global__ void k_f2bf(const float* __restrict__ src,
                       unsigned short* __restrict__ dst, unsigned n) {
  unsigned i = blockIdx.x * blockDim.x + threadIdx.x;
  if (i < n) dst[i] = f2bf(src[i]);
}

__global__ void k_zero_f32(float* __restrict__ p, unsigned n) {
  unsigned i = blockIdx.x * blockDim.x + threadIdx.x;
  if (i < n) p[i] = 0.0f;
}

// ---------------------------------------------------------------------------
// Implicit-GEMM conv / FC kernel.
//   C[M][N] = W[M][K] x Im2col(act)[K][N],  N = BQ*OH*OW, K = Cin*KH*KW
//   act layout: [Cin][BQ*H*W] bf16 bits; stride-1 convs only (all layers are).
// Block: 256 threads = 8 waves (2 M-waves x 4 N-waves); tile 64(M) x 128(N).
// Each wave: 32x32 tile = 4 WMMA accumulators per 32-K chunk.
// denseA != 0 requires M%64==0 && K%32==0: weight tile staged via
// GLOBAL_LOAD_ASYNC_TO_LDS_B128 (no guards needed).
// __launch_bounds__(256, 1): don't cap VGPRs for occupancy -- 4 independent
// WMMA chains/wave already hide latency; avoid scratch spills instead.
// ---------------------------------------------------------------------------
template <int KH, int KW, int PAD>
__launch_bounds__(256, 1)
__global__ void k_conv_wmma(const unsigned short* __restrict__ act,
                            const unsigned short* __restrict__ wgt,
                            const float* __restrict__ bias,
                            unsigned short* __restrict__ outMN,  // bf16 [M][N]
                            float* __restrict__ outNM,           // f32  [N][M]
                            int M, int Cin, int H, int W, int OH, int OW,
                            int doRelu, int outIsNM, int denseA) {
  const unsigned N = (unsigned)BQ * OH * OW;
  const unsigned K = (unsigned)Cin * KH * KW;
  constexpr int khkw = KH * KW;

  // Row stride 40 halves = 80B keeps 16B alignment for b128 LDS accesses.
  __shared__ __align__(16) unsigned short As[64][40];   // [m][k]
  __shared__ __align__(16) unsigned short Bs[128][40];  // [n][k]

  const int t      = threadIdx.x;
  const int lane   = t & 31;
  const int wave   = t >> 5;
  const int wm     = wave >> 2;  // 0..1 -> M offset wm*32
  const int wn     = wave & 3;   // 0..3 -> N offset wn*32
  const int laneHi = lane >> 4;  // 0/1
  const int lane15 = lane & 15;

  const unsigned mBlock = blockIdx.y * 64u;
  const unsigned nBlock = blockIdx.x * 128u;

  // --- per-thread B (im2col) gather coordinates: 2 threads per column ----
  const int      colB  = t >> 1;          // 0..127
  const int      kHalf = (t & 1) * 16;    // 0 or 16
  const unsigned nG    = nBlock + colB;
  const unsigned ohow  = (unsigned)OH * OW;
  const unsigned bImg  = nG / ohow;       // 32-bit division
  const unsigned rr    = nG - bImg * ohow;
  const int      oy    = (int)(rr / (unsigned)OW);
  const int      ox    = (int)(rr - (unsigned)oy * OW);
  const unsigned hw          = (unsigned)H * W;
  const unsigned actImg      = bImg * hw;
  const unsigned planeStride = (unsigned)BQ * hw;

  // --- per-thread A staging coordinates: 4 threads per row, 8 halves -----
  const int rA = t >> 2;        // 0..63
  const int cA = (t & 3) * 8;   // 0,8,16,24
  const unsigned ldsA = (unsigned)(unsigned long long)(uintptr_t)&As[rA][cA];
  const unsigned short* wRow = wgt + (unsigned)(mBlock + rA) * K + cA;

  v8f acc00 = {0, 0, 0, 0, 0, 0, 0, 0};
  v8f acc01 = {0, 0, 0, 0, 0, 0, 0, 0};
  v8f acc10 = {0, 0, 0, 0, 0, 0, 0, 0};
  v8f acc11 = {0, 0, 0, 0, 0, 0, 0, 0};

  for (unsigned k0 = 0; k0 < K; k0 += 32) {
    // ---- stage weight tile 64x32 -------------------------------------
    if (denseA) {
      asm volatile("global_load_async_to_lds_b128 %0, %1, off"
                   :: "v"(ldsA),
                      "v"((unsigned long long)(uintptr_t)(wRow + k0))
                   : "memory");
    } else {
      const unsigned gm = mBlock + rA;
#pragma unroll
      for (int j = 0; j < 8; j++) {
        const unsigned gk = k0 + cA + j;
        unsigned short v = 0;
        if (gm < (unsigned)M && gk < K) v = wRow[k0 + j];
        As[rA][cA + j] = v;
      }
    }
    // ---- stage im2col tile 128x32 (gathered) -------------------------
#pragma unroll 4
    for (int kk = 0; kk < 16; kk++) {
      const unsigned k = k0 + kHalf + kk;
      unsigned short v = 0;
      if (k < K) {
        const unsigned ci = k / (unsigned)khkw;   // constant divisor
        const unsigned f  = k - ci * khkw;
        const int ky = (int)(f / (unsigned)KW);
        const int kx = (int)(f - (unsigned)ky * KW);
        const int y  = oy + ky - PAD;
        const int x  = ox + kx - PAD;
        if (y >= 0 && y < H && x >= 0 && x < W)
          v = act[ci * planeStride + actImg + (unsigned)y * W + x];
      }
      Bs[colB][kHalf + kk] = v;
    }
    if (denseA)
      asm volatile("s_wait_asynccnt 0x0" ::: "memory");
    __syncthreads();

    // ---- fragments: contiguous k order, same for A and B ------------------------------
    //  A 16x32: halves 0..15 -> K = laneHi*16 + h (contiguous 32B)
    //  B 32x16: halves 0..15 -> K = laneHi*16 + h (contiguous 32B)
    const int aRow = wm * 32 + lane15;
    const int aOff = laneHi * 16;
    const v16bf a0 = ldsFrag(&As[aRow][aOff],      &As[aRow][aOff + 8]);
    const v16bf a1 = ldsFrag(&As[aRow + 16][aOff], &As[aRow + 16][aOff + 8]);

    const int bOff = laneHi * 16;
    const int c0   = wn * 32 + lane15;
    const v16bf b0 = ldsFrag(&Bs[c0][bOff],      &Bs[c0][bOff + 8]);
    const v16bf b1 = ldsFrag(&Bs[c0 + 16][bOff], &Bs[c0 + 16][bOff + 8]);

    acc00 = __builtin_amdgcn_wmma_f32_16x16x32_bf16(false, a0, false, b0,
                                                    (short)0, acc00, false, false);
    acc01 = __builtin_amdgcn_wmma_f32_16x16x32_bf16(false, a0, false, b1,
                                                    (short)0, acc01, false, false);
    acc10 = __builtin_amdgcn_wmma_f32_16x16x32_bf16(false, a1, false, b0,
                                                    (short)0, acc10, false, false);
    acc11 = __builtin_amdgcn_wmma_f32_16x16x32_bf16(false, a1, false, b1,
                                                    (short)0, acc11, false, false);
    __syncthreads();
  }

  // ---- epilogue: D layout -> m = e + laneHi*8, n = lane15 --------------
#pragma unroll
  for (int ai = 0; ai < 2; ai++) {
#pragma unroll
    for (int bi = 0; bi < 2; bi++) {
      const v8f a = (ai == 0) ? (bi == 0 ? acc00 : acc01)
                              : (bi == 0 ? acc10 : acc11);
      const unsigned n = nBlock + wn * 32 + bi * 16 + lane15;
#pragma unroll
      for (int e = 0; e < 8; e++) {
        const unsigned m = mBlock + wm * 32 + ai * 16 + e + laneHi * 8;
        if (m < (unsigned)M) {
          float val = a[e] + (bias ? bias[m] : 0.0f);
          if (doRelu) val = fmaxf(val, 0.0f);
          if (outIsNM)
            outNM[n * (unsigned)M + m] = val;
          else
            outMN[m * N + n] = f2bf(val);
        }
      }
    }
  }
}

// ---------------------------------------------------------------------------
// maxpool on [C][BQ*H*W] bf16 -> [C][BQ*OH*OW]
// ---------------------------------------------------------------------------
__global__ void k_maxpool(const unsigned short* __restrict__ in,
                          unsigned short* __restrict__ out,
                          int C, int H, int W, int OH, int OW,
                          int KS, int ST, int PAD) {
  const unsigned total = (unsigned)C * BQ * OH * OW;
  unsigned idx = blockIdx.x * blockDim.x + threadIdx.x;
  if (idx >= total) return;
  unsigned r = idx;
  const int ox = (int)(r % (unsigned)OW); r /= (unsigned)OW;
  const int oy = (int)(r % (unsigned)OH); r /= (unsigned)OH;
  const int b  = (int)(r % BQ);           r /= BQ;
  const int c  = (int)r;
  const unsigned base = ((unsigned)c * BQ + b) * (unsigned)(H * W);
  float m = -INFINITY;
  for (int ky = 0; ky < KS; ky++) {
    const int y = oy * ST - PAD + ky;
    if (y < 0 || y >= H) continue;
    for (int kx = 0; kx < KS; kx++) {
      const int x = ox * ST - PAD + kx;
      if (x < 0 || x >= W) continue;
      m = fmaxf(m, bf2f(in[base + (unsigned)y * W + x]));
    }
  }
  out[idx] = f2bf(m);
}

// ---------------------------------------------------------------------------
// PrimaryCaps squash: pc [256][BQ*16] bf16 -> u [BQ][512][8] f32
// ---------------------------------------------------------------------------
__global__ void k_squash_primary(const unsigned short* __restrict__ pc,
                                 float* __restrict__ u) {
  const unsigned idx = blockIdx.x * blockDim.x + threadIdx.x;  // b*512 + i
  if (idx >= BQ * 512) return;
  const unsigned b = idx >> 9;
  const unsigned i = idx & 511;
  float d8[8], sq = 0.0f;
#pragma unroll
  for (unsigned d = 0; d < 8; d++) {
    const unsigned j = i * 8 + d;
    const unsigned c = j >> 4;
    const unsigned s = j & 15;
    const float v = bf2f(pc[c * (BQ * 16) + b * 16 + s]);
    d8[d] = v;
    sq += v * v;
  }
  const float scale = (sq / (1.0f + sq)) * rsqrtf(sq + 1e-8f);
#pragma unroll
  for (unsigned d = 0; d < 8; d++) u[idx * 8 + d] = d8[d] * scale;
}

// ---------------------------------------------------------------------------
// x_hat[b,o,i,v] = sum_d Wr[o,i,v,d] * u[b,i,d]
// ---------------------------------------------------------------------------
__global__ void k_xhat(const float* __restrict__ Wr,
                       const float* __restrict__ u,
                       float* __restrict__ xh) {
  const unsigned idx = blockIdx.x * blockDim.x + threadIdx.x;
  if (idx >= (unsigned)BQ * 10 * 512) return;
  const unsigned i = idx & 511;
  const unsigned r = idx >> 9;
  const unsigned o = r % 10;
  const unsigned b = r / 10;
  const float* up = u + (b * 512 + i) * 8;
  float ud[8];
#pragma unroll
  for (int d = 0; d < 8; d++) ud[d] = up[d];
  const float* wp = Wr + (o * 512 + i) * 128;
  float* xp = xh + (size_t)idx * 16;
#pragma unroll
  for (int v = 0; v < 16; v++) {
    float s = 0.0f;
#pragma unroll
    for (int d = 0; d < 8; d++) s += wp[v * 8 + d] * ud[d];
    xp[v] = s;
  }
}

// ---------------------------------------------------------------------------
// one dynamic-routing iteration; block per batch item, 256 threads.
//  blog [BQ][10][512], xh [BQ][10][512][16]
//  last iter writes f (bf16, feature-major [160][BQ]) for the FC head.
// ---------------------------------------------------------------------------
__global__ void k_route(const float* __restrict__ xh,
                        float* __restrict__ blog,
                        unsigned short* __restrict__ f, int isLast) {
  const int b = blockIdx.x;
  const int t = threadIdx.x;
  __shared__ float cs[10][512];
  __shared__ float vs[10][16];
  const float* xb = xh + (size_t)b * 10 * 512 * 16;
  float* bb = blog + (size_t)b * 10 * 512;

  // softmax over o=10 for each capsule i
  for (int ii = t; ii < 512; ii += 256) {
    float bl[10], mx = -INFINITY;
#pragma unroll
    for (int o = 0; o < 10; o++) { bl[o] = bb[o * 512 + ii]; mx = fmaxf(mx, bl[o]); }
    float sum = 0.0f;
#pragma unroll
    for (int o = 0; o < 10; o++) { bl[o] = expf(bl[o] - mx); sum += bl[o]; }
    const float inv = 1.0f / sum;
#pragma unroll
    for (int o = 0; o < 10; o++) cs[o][ii] = bl[o] * inv;
  }
  __syncthreads();

  // v[o][vd] = sum_i c[o][i] * xh[o][i][vd]
  if (t < 160) {
    const int o = t >> 4, v = t & 15;
    float s = 0.0f;
    for (int i = 0; i < 512; i++) s += cs[o][i] * xb[(unsigned)(o * 512 + i) * 16 + v];
    vs[o][v] = s;
  }
  __syncthreads();
  if (t < 10) {
    float sq = 0.0f;
#pragma unroll
    for (int v = 0; v < 16; v++) { const float x = vs[t][v]; sq += x * x; }
    const float scale = (sq / (1.0f + sq)) * rsqrtf(sq + 1e-8f);
#pragma unroll
    for (int v = 0; v < 16; v++) vs[t][v] *= scale;
  }
  __syncthreads();

  if (isLast) {
    if (t < 160) {
      const int o = t >> 4, v = t & 15;
      f[(unsigned)(o * 16 + v) * BQ + b] = f2bf(vs[o][v]);
    }
  } else {
    for (int ii = t; ii < 512; ii += 256) {
      for (int o = 0; o < 10; o++) {
        const float* xp = xb + (unsigned)(o * 512 + ii) * 16;
        float dot = 0.0f;
#pragma unroll
        for (int v = 0; v < 16; v++) dot += vs[o][v] * xp[v];
        bb[o * 512 + ii] += dot;
      }
    }
  }
}

// ---------------------------------------------------------------------------
// host launcher
// ---------------------------------------------------------------------------
extern "C" void kernel_launch(void* const* d_in, const int* in_sizes, int n_in,
                              void* d_out, int out_size, void* d_ws, size_t ws_size,
                              hipStream_t stream) {
  const float* x   = (const float*)d_in[0];
  const float* w1  = (const float*)d_in[1];
  const float* b1  = (const float*)d_in[2];
  const float* w2  = (const float*)d_in[3];
  const float* b2  = (const float*)d_in[4];
  const float* w3  = (const float*)d_in[5];
  const float* b3  = (const float*)d_in[6];
  const float* w4  = (const float*)d_in[7];
  const float* b4  = (const float*)d_in[8];
  const float* w5  = (const float*)d_in[9];
  const float* b5  = (const float*)d_in[10];
  const float* wp  = (const float*)d_in[11];
  const float* bp  = (const float*)d_in[12];
  const float* Wr  = (const float*)d_in[13];
  const float* fw1 = (const float*)d_in[14];
  const float* fb1 = (const float*)d_in[15];
  const float* fw2 = (const float*)d_in[16];
  const float* fb2 = (const float*)d_in[17];
  const float* fw3 = (const float*)d_in[18];
  const float* fb3 = (const float*)d_in[19];
  float* out = (float*)d_out;

  char* ws = (char*)d_ws;
  size_t off = 0;
  auto alloc = [&](size_t bytes) -> char* {
    off = (off + 255) & ~(size_t)255;
    char* p = ws + off;
    off += bytes;
    return p;
  };
  auto aU16 = [&](size_t n) -> unsigned short* { return (unsigned short*)alloc(n * 2); };
  auto aF32 = [&](size_t n) -> float* { return (float*)alloc(n * 4); };

  auto cvt = [&](const float* s, unsigned short* d, unsigned n) {
    k_f2bf<<<dim3((n + 255) / 256), dim3(256), 0, stream>>>(s, d, n);
  };
  // kind: 0 -> 3x3 pad1, 1 -> 2x2 pad0, 2 -> 1x1 pad0
  auto conv = [&](int kind, const unsigned short* act, const unsigned short* wgt,
                  const float* bias, unsigned short* omn, float* onm,
                  int M, int Cin, int H, int W, int OH, int OW,
                  int relu, int nm, int denseA) {
    dim3 grid((unsigned)((BQ * OH * OW) / 128), (unsigned)((M + 63) / 64));
    if (kind == 0)
      k_conv_wmma<3, 3, 1><<<grid, dim3(256), 0, stream>>>(
          act, wgt, bias, omn, onm, M, Cin, H, W, OH, OW, relu, nm, denseA);
    else if (kind == 1)
      k_conv_wmma<2, 2, 0><<<grid, dim3(256), 0, stream>>>(
          act, wgt, bias, omn, onm, M, Cin, H, W, OH, OW, relu, nm, denseA);
    else
      k_conv_wmma<1, 1, 0><<<grid, dim3(256), 0, stream>>>(
          act, wgt, bias, omn, onm, M, Cin, H, W, OH, OW, relu, nm, denseA);
  };
  auto pool = [&](const unsigned short* in, unsigned short* o,
                  int C, int H, int W, int OH, int OW, int KS, int ST, int PAD) {
    unsigned n = (unsigned)C * BQ * OH * OW;
    k_maxpool<<<dim3((n + 255) / 256), dim3(256), 0, stream>>>(
        in, o, C, H, W, OH, OW, KS, ST, PAD);
  };

  // --- workspace layout --------------------------------------------------
  unsigned short* xbf  = aU16(in_sizes[0]);
  unsigned short* w1b  = aU16(in_sizes[1]);
  unsigned short* w2b  = aU16(in_sizes[3]);
  unsigned short* w3b  = aU16(in_sizes[5]);
  unsigned short* w4b  = aU16(in_sizes[7]);
  unsigned short* w5b  = aU16(in_sizes[9]);
  unsigned short* wpb  = aU16(in_sizes[11]);
  unsigned short* fw1b = aU16(in_sizes[14]);
  unsigned short* fw2b = aU16(in_sizes[16]);
  unsigned short* fw3b = aU16(in_sizes[18]);

  unsigned short* a1 = aU16((size_t)96  * BQ * 28 * 28);
  unsigned short* p1 = aU16((size_t)96  * BQ * 14 * 14);
  unsigned short* a2 = aU16((size_t)256 * BQ * 14 * 14);
  unsigned short* p2 = aU16((size_t)256 * BQ * 7 * 7);
  unsigned short* a3 = aU16((size_t)384 * BQ * 7 * 7);
  unsigned short* a4 = aU16((size_t)384 * BQ * 7 * 7);
  unsigned short* a5 = aU16((size_t)256 * BQ * 7 * 7);
  unsigned short* p3 = aU16((size_t)256 * BQ * 5 * 5);
  unsigned short* pc = aU16((size_t)256 * BQ * 4 * 4);
  float* u    = aF32((size_t)BQ * 512 * 8);
  float* xh   = aF32((size_t)BQ * 10 * 512 * 16);
  float* blog = aF32((size_t)BQ * 10 * 512);
  unsigned short* fcaps = aU16((size_t)160 * BQ);
  unsigned short* fc1o  = aU16((size_t)4096 * BQ);
  unsigned short* fc2o  = aU16((size_t)4096 * BQ);
  (void)ws_size; (void)n_in; (void)out_size;

  // --- bf16 conversions --------------------------------------------------
  cvt(x, xbf, in_sizes[0]);
  cvt(w1, w1b, in_sizes[1]);    cvt(w2, w2b, in_sizes[3]);
  cvt(w3, w3b, in_sizes[5]);    cvt(w4, w4b, in_sizes[7]);
  cvt(w5, w5b, in_sizes[9]);    cvt(wp, wpb, in_sizes[11]);
  cvt(fw1, fw1b, in_sizes[14]); cvt(fw2, fw2b, in_sizes[16]);
  cvt(fw3, fw3b, in_sizes[18]);

  // --- conv stack ---------------------------------------------------------
  conv(0, xbf, w1b, b1, a1, nullptr,  96,   1, 28, 28, 28, 28, 1, 0, 0); // K=9: guarded path
  pool(a1, p1, 96, 28, 28, 14, 14, 3, 2, 1);
  conv(0, p1, w2b, b2, a2, nullptr, 256,  96, 14, 14, 14, 14, 1, 0, 1);
  pool(a2, p2, 256, 14, 14, 7, 7, 3, 2, 1);
  conv(0, p2, w3b, b3, a3, nullptr, 384, 256,  7,  7,  7,  7, 1, 0, 1);
  conv(0, a3, w4b, b4, a4, nullptr, 384, 384,  7,  7,  7,  7, 1, 0, 1);
  conv(0, a4, w5b, b5, a5, nullptr, 256, 384,  7,  7,  7,  7, 1, 0, 1);
  pool(a5, p3, 256, 7, 7, 5, 5, 3, 1, 0);
  conv(1, p3, wpb, bp, pc, nullptr, 256, 256,  5,  5,  4,  4, 0, 0, 1);  // no relu

  // --- capsules + routing -------------------------------------------------
  k_squash_primary<<<dim3((BQ * 512 + 255) / 256), dim3(256), 0, stream>>>(pc, u);
  {
    unsigned n = (unsigned)BQ * 10 * 512;
    k_xhat<<<dim3((n + 255) / 256), dim3(256), 0, stream>>>(Wr, u, xh);
    k_zero_f32<<<dim3((n + 255) / 256), dim3(256), 0, stream>>>(blog, n);
  }
  k_route<<<dim3(BQ), dim3(256), 0, stream>>>(xh, blog, fcaps, 0);
  k_route<<<dim3(BQ), dim3(256), 0, stream>>>(xh, blog, fcaps, 0);
  k_route<<<dim3(BQ), dim3(256), 0, stream>>>(xh, blog, fcaps, 1);

  // --- FC head (feature-major activations [K][BQ]) -------------------------
  conv(2, fcaps, fw1b, fb1, fc1o, nullptr, 4096,  160, 1, 1, 1, 1, 1, 0, 1);
  conv(2, fc1o,  fw2b, fb2, fc2o, nullptr, 4096, 4096, 1, 1, 1, 1, 1, 0, 1);
  conv(2, fc2o,  fw3b, fb3, nullptr, out,    10, 4096, 1, 1, 1, 1, 0, 1, 0); // f32 [B][10]
}
